// LocalAttention_41858751267186
// MI455X (gfx1250) — matrix operand order repaired and checked by reading, not compiled, not run on hardware
//
#include <hip/hip_runtime.h>

#define DIMX   128
#define NHEAD  8
#define NNB    32
#define CDIMC  1024
#define BC     16384          // B*C = 16*1024
#define PITCH  136            // halves per LDS row (272B -> bank-rotating)

typedef _Float16 h8    __attribute__((ext_vector_type(8)));
typedef _Float16 v16h  __attribute__((ext_vector_type(16)));
typedef float    v8f   __attribute__((ext_vector_type(8)));
typedef unsigned int u32x4 __attribute__((ext_vector_type(4)));
typedef int      i32x8 __attribute__((ext_vector_type(8)));
typedef int      i32x4 __attribute__((ext_vector_type(4)));

#if __has_builtin(__builtin_amdgcn_tensor_load_to_lds)
#define LA_HAVE_TDM 1
#else
#define LA_HAVE_TDM 0
#endif

// LDS partition byte offsets (dynamic LDS)
#define OFF_WK    0u        // 128*136 halves = 34816 B
#define OFF_WV    34816u    // 34816 B
#define OFF_A     69632u    // 32*136 halves = 8704 B
#define OFF_RAW   78336u    // 32*128 f32 = 16384 B
#define OFF_AQ    94720u    // 2 x 128 f32 (double buffered)
#define OFF_WD    95744u    // 128 f32
#define OFF_BD    96256u    // 128 f32
#define OFF_DIST  96768u    // 2 x 32 f32 (double buffered)
#define OFF_ATTN  97024u    // 8*32 f32
#define OFF_Q     98048u    // 128 f32 (scaled q)
#define SMEM_BYTES 98560u

// --- WMMA fragment loaders (layouts per CDNA5 ISA 7.12.2, 16-bit A 16x32 / B 32x16) ---
// A: lane l<16 holds row M=l%16, K halves {0..7,16..23}; lane>=16 holds {8..15,24..31}
__device__ __forceinline__ v16h load_fragA(const _Float16* p, int row, int s, int lane) {
    int base = row * PITCH + s * 32 + ((lane & 16) ? 8 : 0);
    h8 lo = *(const h8*)(p + base);
    h8 hi = *(const h8*)(p + base + 16);
    return __builtin_shufflevector(lo, hi, 0,1,2,3,4,5,6,7,8,9,10,11,12,13,14,15);
}
// B: lane l holds col N=l%16, K halves {0..7,16..23} for l<16; {8..15,24..31} for l>=16
__device__ __forceinline__ v16h load_fragB(const _Float16* p, int ncol, int s, int lane) {
    int base = ncol * PITCH + s * 32 + ((lane & 16) ? 8 : 0);
    h8 lo = *(const h8*)(p + base);
    h8 hi = *(const h8*)(p + base + 16);
    return __builtin_shufflevector(lo, hi, 0,1,2,3,4,5,6,7,8,9,10,11,12,13,14,15);
}

#if LA_HAVE_TDM
// Issue 1-D TDM copy: 16 KB neighbor slab (32x128 f32) global -> LDS (async, TENSORcnt)
__device__ __forceinline__ void tdm_issue_slab(const float* gsrc, unsigned int lds_byte_addr) {
    unsigned long long ga = (unsigned long long)(uintptr_t)gsrc;
    u32x4 g0; g0[0] = 1u;                       // count=1, valid user descriptor
    g0[1] = lds_byte_addr;                      // lds_addr
    g0[2] = (unsigned int)ga;                   // global_addr[31:0]
    g0[3] = (unsigned int)((ga >> 32) & 0x1FFFFFFu) | 0x80000000u; // addr hi | type=2
    i32x8 g1;
    g1[0] = 0x00020000;  // data_size=2 (4B elements), no multicast/pad
    g1[1] = 0x10000000;  // tensor_dim0[15:0]=4096 at bits[63:48]
    g1[2] = 0x00010000;  // tensor_dim1=1
    g1[3] = 0x10000000;  // tile_dim0=4096 at bits[127:112]
    g1[4] = 0;           // tile_dim1/2 unused
    g1[5] = 4096;        // tensor_dim0_stride
    g1[6] = 0; g1[7] = 0;
    i32x4 z; z[0] = z[1] = z[2] = z[3] = 0;
    i32x8 z8; z8[0]=z8[1]=z8[2]=z8[3]=z8[4]=z8[5]=z8[6]=z8[7]=0;
    __builtin_amdgcn_tensor_load_to_lds(g0, g1, z, z, z8, 0);
}
#endif

// Convert f32 row-major weights [k][n] -> transposed f16 [n][k] in workspace
__global__ void la_prep_kernel(const float* __restrict__ Wq, const float* __restrict__ Wk,
                               const float* __restrict__ Wv,
                               _Float16* __restrict__ WqT, _Float16* __restrict__ WkT,
                               _Float16* __restrict__ WvT) {
    int n = blockIdx.x, k = threadIdx.x;
    WqT[n * DIMX + k] = (_Float16)Wq[k * DIMX + n];
    WkT[n * DIMX + k] = (_Float16)Wk[k * DIMX + n];
    WvT[n * DIMX + k] = (_Float16)Wv[k * DIMX + n];
}

__global__ __launch_bounds__(256)
void la_main_kernel(const float* __restrict__ aq_g, const float* __restrict__ an_g,
                    const float* __restrict__ dist_g, const float* __restrict__ mask_g,
                    const float* __restrict__ Wd_g, const float* __restrict__ bd_g,
                    const float* __restrict__ bq_g, const float* __restrict__ bk_g,
                    const float* __restrict__ bv_g,
                    const _Float16* __restrict__ WqT, const _Float16* __restrict__ WkT,
                    const _Float16* __restrict__ WvT,
                    float* __restrict__ out_attn, float* __restrict__ out_ctx) {
    extern __shared__ __align__(16) char smem[];
    _Float16* sWk  = (_Float16*)(smem + OFF_WK);
    _Float16* sWv  = (_Float16*)(smem + OFF_WV);
    _Float16* sA   = (_Float16*)(smem + OFF_A);
    float*    sRaw = (float*)(smem + OFF_RAW);
    float*    sAQ  = (float*)(smem + OFF_AQ);   // [2][128]
    float*    sWd  = (float*)(smem + OFF_WD);
    float*    sbd  = (float*)(smem + OFF_BD);
    float*    sDist= (float*)(smem + OFF_DIST); // [2][32]
    float*    sAttn= (float*)(smem + OFF_ATTN);
    float*    sQ   = (float*)(smem + OFF_Q);

    const int tid  = threadIdx.x;
    const int wv   = tid >> 5;          // wave id == head id
    const int lane = tid & 31;
    const int lo16 = lane & 15;
    const int hi   = lane >> 4;         // 0/1
    const int d    = wv * 16 + lo16;    // column / head-dim owned by this lane

    // ---- prologue: kick off unit0's neighbor DMA before anything else ----
    const int unit0 = blockIdx.x;
#if LA_HAVE_TDM
    if (wv == 0)
        tdm_issue_slab(an_g + (size_t)unit0 * (NNB * DIMX), (unsigned int)(uintptr_t)sRaw);
#else
    for (int i = tid; i < NNB * DIMX; i += 256)
        sRaw[i] = an_g[(size_t)unit0 * (NNB * DIMX) + i];
#endif
    // stage f16 weights into LDS once per block (pitched), overlapped with the DMA
    for (int i = tid; i < 2048; i += 256) {          // 16B chunks, 128 rows x 16 chunks
        int n  = i >> 4;
        int k8 = (i & 15) << 3;
        *(h8*)(sWk + n * PITCH + k8) = *(const h8*)(WkT + n * DIMX + k8);
        *(h8*)(sWv + n * PITCH + k8) = *(const h8*)(WvT + n * DIMX + k8);
    }
    if (tid < DIMX) { sWd[tid] = Wd_g[tid]; sbd[tid] = bd_g[tid]; }
    if (tid < DIMX) sAQ[tid]   = aq_g[(size_t)unit0 * DIMX + tid];
    if (tid < NNB)  sDist[tid] = dist_g[(size_t)unit0 * NNB + tid];
    __syncthreads();

    int buf = 0;
    for (int unit = blockIdx.x; unit < BC; unit += gridDim.x) {
        const int b = unit >> 10;
        const int c = unit & (CDIMC - 1);
        const float* sAQc = sAQ + buf * DIMX;
        const float* sDc  = sDist + buf * NNB;

#if LA_HAVE_TDM
        if (wv == 0) __builtin_amdgcn_s_wait_tensorcnt(0);   // sRaw for this unit ready
#endif
        __syncthreads();

        // ---- distance gate (swish) + f32->f16, write pitched A matrix ----
        #pragma unroll 4
        for (int i = tid; i < NNB * DIMX; i += 256) {
            int n  = i >> 7;
            int dd = i & (DIMX - 1);
            float x = sDc[n] * sWd[dd] + sbd[dd];
            float g = x / (1.0f + __expf(-x));          // swish
            sA[n * PITCH + dd] = (_Float16)(sRaw[i] * g);
        }
        __syncthreads();                                 // sA ready; sRaw free

        // ---- prefetch next unit: async DMA + small stages into the other buffer ----
        const int nunit = unit + gridDim.x;
        if (nunit < BC) {
#if LA_HAVE_TDM
            if (wv == 0)
                tdm_issue_slab(an_g + (size_t)nunit * (NNB * DIMX),
                               (unsigned int)(uintptr_t)sRaw);
#else
            for (int i = tid; i < NNB * DIMX; i += 256)
                sRaw[i] = an_g[(size_t)nunit * (NNB * DIMX) + i];
#endif
            if (tid < DIMX) sAQ[(buf ^ 1) * DIMX + tid] = aq_g[(size_t)nunit * DIMX + tid];
            if (tid < NNB)  sDist[(buf ^ 1) * NNB + tid] = dist_g[(size_t)nunit * NNB + tid];
        }

        // ---- Q projection for this wave's 16 columns (VALU, K=128 split across lane halves) ----
        float qacc = 0.0f;
        {
            const _Float16* wqrow = WqT + d * DIMX + hi * 64;
            const float*    aqh   = sAQc + hi * 64;
            #pragma unroll
            for (int kk = 0; kk < 64; kk += 8) {
                h8 wl = *(const h8*)(wqrow + kk);
                #pragma unroll
                for (int e = 0; e < 8; ++e) qacc += aqh[kk + e] * (float)wl[e];
            }
        }
        qacc += __shfl_xor(qacc, 16);
        const float qs = (qacc + bq_g[d]) * 0.25f;      // * HDIM^-0.5
        if (hi == 0) sQ[d] = qs;                        // publish q for row-indexed use

        // per-lane row constants: rows this lane holds are d' = wv*16 + hi*8 + j
        float qh[8], bkh[8];
        #pragma unroll
        for (int j = 0; j < 8; ++j) {
            qh[j]  = sQ[wv * 16 + hi * 8 + j];
            bkh[j] = bk_g[wv * 16 + hi * 8 + j];
        }

        // ---- Kt = WkT @ an^T (transposed K GEMM): D-tile rows = head dims, cols = neighbors.
        //      energy[n] = sum_d q[d]*k[n][d] reduces across VGPRs (pure VALU) + one xor16. ----
        v16h ak[4];
        #pragma unroll
        for (int s = 0; s < 4; ++s) ak[s] = load_fragA(sWk, d, s, lane);

        float e01[2];
        #pragma unroll
        for (int m = 0; m < 2; ++m) {
            v8f acc;
            #pragma unroll
            for (int j = 0; j < 8; ++j) acc[j] = bkh[j];   // bk bias per d-row
            #pragma unroll
            for (int s = 0; s < 4; ++s) {
                v16h bm = load_fragB(sA, m * 16 + lo16, s, lane);
                acc = __builtin_amdgcn_wmma_f32_16x16x32_f16(
                        false, ak[s], false, bm, (short)0, acc, false, false);
            }
            float ep = 0.0f;
            #pragma unroll
            for (int j = 0; j < 8; ++j) ep += qh[j] * acc[j];
            e01[m] = ep + __shfl_xor(ep, 16);           // all lanes: energy[m*16 + lo16]
        }
        float e = hi ? e01[1] : e01[0];                 // lane n holds energy[n]

        // ---- softmax over 32 neighbors (wave-local; lane == neighbor) ----
        {
            float mval = mask_g[(size_t)unit * NNB + lane];
            e += (1.0f - mval) * (-1.0e9f);
            float mx = e;
            #pragma unroll
            for (int o = 16; o >= 1; o >>= 1) mx = fmaxf(mx, __shfl_xor(mx, o));
            float ex = __expf(e - mx);
            float sm = ex;
            #pragma unroll
            for (int o = 16; o >= 1; o >>= 1) sm += __shfl_xor(sm, o);
            float at = ex / sm;
            out_attn[(((size_t)b * NHEAD + wv) * CDIMC + c) * NNB + lane] = at;
            sAttn[wv * 32 + lane] = at * mval;          // masked attn for context
        }

        // ---- V = gated_neighbors @ Wv (+bv), fused attn-weighted sum ----
        const float bvd = bv_g[d];
        v16h bv4[4];
        #pragma unroll
        for (int s = 0; s < 4; ++s) bv4[s] = load_fragB(sWv, d, s, lane);

        float ctx = 0.0f;
        #pragma unroll
        for (int m = 0; m < 2; ++m) {
            v8f acc;
            #pragma unroll
            for (int j = 0; j < 8; ++j) acc[j] = bvd;
            #pragma unroll
            for (int s = 0; s < 4; ++s) {
                v16h a = load_fragA(sA, m * 16 + lo16, s, lane);
                acc = __builtin_amdgcn_wmma_f32_16x16x32_f16(
                        false, a, false, bv4[s], (short)0, acc, false, false);
            }
            #pragma unroll
            for (int j = 0; j < 8; ++j)
                ctx += sAttn[wv * 32 + m * 16 + hi * 8 + j] * acc[j];
        }
        ctx += __shfl_xor(ctx, 16);                     // combine row halves (same column)
        if (hi == 0) out_ctx[(size_t)unit * DIMX + d] = ctx + sAQc[d];  // +residual

        buf ^= 1;
    }
}

extern "C" void kernel_launch(void* const* d_in, const int* in_sizes, int n_in,
                              void* d_out, int out_size, void* d_ws, size_t ws_size,
                              hipStream_t stream) {
    const float* atom_query    = (const float*)d_in[0];
    const float* atom_neighbor = (const float*)d_in[1];
    const float* ndist         = (const float*)d_in[2];
    const float* mask          = (const float*)d_in[3];
    const float* Wd            = (const float*)d_in[4];
    const float* bd            = (const float*)d_in[5];
    const float* Wq            = (const float*)d_in[6];
    const float* bq            = (const float*)d_in[7];
    const float* Wk            = (const float*)d_in[8];
    const float* bk            = (const float*)d_in[9];
    const float* Wv            = (const float*)d_in[10];
    const float* bv            = (const float*)d_in[11];

    float* out_attn = (float*)d_out;                       // [16,8,1024,32]
    float* out_ctx  = out_attn + (size_t)16 * NHEAD * CDIMC * NNB;  // [16,1024,128]

    _Float16* WqT = (_Float16*)d_ws;
    _Float16* WkT = WqT + DIMX * DIMX;
    _Float16* WvT = WkT + DIMX * DIMX;

    la_prep_kernel<<<DIMX, DIMX, 0, stream>>>(Wq, Wk, Wv, WqT, WkT, WvT);

    (void)hipFuncSetAttribute((const void*)la_main_kernel,
                              hipFuncAttributeMaxDynamicSharedMemorySize,
                              (int)SMEM_BYTES);
    la_main_kernel<<<2048, 256, SMEM_BYTES, stream>>>(
        atom_query, atom_neighbor, ndist, mask, Wd, bd, bq, bk, bv,
        WqT, WkT, WvT, out_attn, out_ctx);
}
